// LogitsAndAdaptiveSoftmax_81028853006378
// MI455X (gfx1250) — matrix operand order repaired and checked by reading, not compiled, not run on hardware
//
#include <hip/hip_runtime.h>

typedef __attribute__((ext_vector_type(16))) __bf16 v16bf;
typedef __attribute__((ext_vector_type(8)))  __bf16 v8bf;
typedef __attribute__((ext_vector_type(4)))  __bf16 v4bf;
typedef __attribute__((ext_vector_type(8)))  float  v8f;

#define BM 128
#define BN 128
#define BK 32
#define BKP 40   // padded LDS row stride (elements): 80B/row = 20 dwords -> conflict-free
                 // lane strides for b128 frag loads, still 16B-aligned everywhere.

// Generic tiled GEMM: C[M,N] (ldc) = A[M,K] (lda) * B[K,N] (ldb), f32 in/out,
// bf16 WMMA with f32 accumulate, double-buffered LDS.
// mode: 0 = no mask, 1 = rows with 2000<=target<10000, 2 = 10000<=target<50000.
// Requires: M % 128 == 0, K % 32 == 0, lda % 4 == 0. N, ldb arbitrary.
__global__ __launch_bounds__(256)
void adasoft_gemm_bf16_wmma(const float* __restrict__ A, int lda,
                            const float* __restrict__ B, int ldb,
                            float* __restrict__ C, int ldc,
                            int M, int N, int K,
                            const int* __restrict__ target, int mode) {
    __shared__ __align__(16) __bf16 lsA[2][BM * BKP];   // [m][k], 10 KB each
    __shared__ __align__(16) __bf16 lsB[2][BN * BKP];   // [n][k] (B transposed), 10 KB each

    const int tid   = threadIdx.x;
    const int lane  = tid & 31;
    const int wave  = tid >> 5;      // 0..7
    const int waveM = wave >> 1;     // 0..3 : 32-row slab
    const int waveN = wave & 1;      // 0..1 : 64-col slab
    const int half  = lane >> 4;     // ISA fragment lane-half split
    const int l16   = lane & 15;

    const int tileM = blockIdx.y * BM;
    const int tileN = blockIdx.x * BN;
    const bool edgeN = (tileN + BN) > N;
    const bool vecB  = ((ldb & 3) == 0) && !edgeN;   // float4-safe B rows

    // A staging: float4 f = tid + i*256 -> row (tid>>3)+i*32, col (tid&7)*4  (coalesced in k)
    const int aR0 = tid >> 3, aC = (tid & 7) * 4;
    const float* aBase = A + (long long)(tileM + aR0) * lda + aC;
    // B staging: float4 f = tid + i*256 -> row (tid>>5)+i*8, col (tid&31)*4  (coalesced in n)
    const int bR0 = tid >> 5, bC = (tid & 31) * 4;
    const float* bBase = B + (long long)bR0 * ldb + (tileN + bC);

    auto loadTile = [&](int kk, float4* aReg, float4* bReg) {
#pragma unroll
        for (int i = 0; i < 4; ++i)
            aReg[i] = *(const float4*)(aBase + (long long)i * 32 * lda + kk);
        if (vecB) {
#pragma unroll
            for (int i = 0; i < 4; ++i)
                bReg[i] = *(const float4*)(bBase + (long long)(kk + i * 8) * ldb);
        } else {
            int gn = tileN + bC;
#pragma unroll
            for (int i = 0; i < 4; ++i) {
                const float* p = bBase + (long long)(kk + i * 8) * ldb;
                bReg[i].x = (gn + 0 < N) ? p[0] : 0.0f;
                bReg[i].y = (gn + 1 < N) ? p[1] : 0.0f;
                bReg[i].z = (gn + 2 < N) ? p[2] : 0.0f;
                bReg[i].w = (gn + 3 < N) ? p[3] : 0.0f;
            }
        }
    };
    auto storeTile = [&](int pb, const float4* aReg, const float4* bReg) {
#pragma unroll
        for (int i = 0; i < 4; ++i) {   // A: packed 4x bf16 = one ds_store_b64
            v4bf a4 = { (__bf16)aReg[i].x, (__bf16)aReg[i].y,
                        (__bf16)aReg[i].z, (__bf16)aReg[i].w };
            *(v4bf*)&lsA[pb][(aR0 + i * 32) * BKP + aC] = a4;
        }
#pragma unroll
        for (int i = 0; i < 4; ++i) {   // B: transpose -> scalar b16 stores
            int r = bR0 + i * 8;
            lsB[pb][(bC + 0) * BKP + r] = (__bf16)bReg[i].x;
            lsB[pb][(bC + 1) * BKP + r] = (__bf16)bReg[i].y;
            lsB[pb][(bC + 2) * BKP + r] = (__bf16)bReg[i].z;
            lsB[pb][(bC + 3) * BKP + r] = (__bf16)bReg[i].w;
        }
    };

    v8f acc[2][4];
#pragma unroll
    for (int i = 0; i < 2; ++i)
#pragma unroll
        for (int j = 0; j < 4; ++j) acc[i][j] = (v8f){0,0,0,0,0,0,0,0};

    // Prologue: stage K-tile 0 into buffer 0.
    {
        float4 aReg[4], bReg[4];
        loadTile(0, aReg, bReg);
        storeTile(0, aReg, bReg);
    }
    __syncthreads();

    int p = 0;
    for (int k0 = 0; k0 < K; k0 += BK) {
        const bool hasNext = (k0 + BK) < K;
        float4 aReg[4], bReg[4];
        // Issue next tile's global loads first: they overlap the WMMA block below.
        if (hasNext) loadTile(k0 + BK, aReg, bReg);
        // Speculative GL2 prefetch of the tile after that.
        if (k0 + 2 * BK < K) {
            __builtin_prefetch(aBase + (long long)0 * lda + (k0 + 2 * BK), 0, 1);
            if (tileN + bC < N)
                __builtin_prefetch(bBase + (long long)(k0 + 2 * BK) * ldb, 0, 1);
        }

        // ---- Fragments from LDS buffer p ----
        // A 16-bit 16x32: lanes 0-15 hold K 0-7 & 16-23 of row l16; lanes 16-31 K 8-15 & 24-31.
        v16bf afrag[2];
#pragma unroll
        for (int i = 0; i < 2; ++i) {
            int m = waveM * 32 + i * 16 + l16;
            const __bf16* q = &lsA[p][m * BKP + half * 8];
            v8bf lo = *(const v8bf*)(q);
            v8bf hi = *(const v8bf*)(q + 16);
            afrag[i] = __builtin_shufflevector(lo, hi, 0,1,2,3,4,5,6,7,8,9,10,11,12,13,14,15);
        }
        // B 32x16: lane = column, lanes 0-15 K 0-7 & 16-23, lanes 16-31 K 8-15 & 24-31 (matches A).
        v16bf bfrag[4];
#pragma unroll
        for (int j = 0; j < 4; ++j) {
            int n = waveN * 64 + j * 16 + l16;
            const __bf16* q = &lsB[p][n * BKP + half * 8];
            v8bf lo = *(const v8bf*)(q);
            v8bf hi = *(const v8bf*)(q + 16);
            bfrag[j] = __builtin_shufflevector(lo, hi, 0,1,2,3,4,5,6,7,8,9,10,11,12,13,14,15);
        }

        // ---- 8 WMMAs: 32x64 sub-tile per wave ----
#pragma unroll
        for (int i = 0; i < 2; ++i)
#pragma unroll
            for (int j = 0; j < 4; ++j)
                acc[i][j] = __builtin_amdgcn_wmma_f32_16x16x32_bf16(
                    false, afrag[i], false, bfrag[j],
                    (short)0, acc[i][j], false, false);

        // Convert + store next tile into the other buffer, then one barrier.
        if (hasNext) storeTile(p ^ 1, aReg, bReg);
        __syncthreads();
        p ^= 1;
    }

    // ---- Epilogue: per-row cluster mask, bounds-checked f32 stores ----
    // C/D layout: lane = col (l16); VGPR r = row r + half*8 within the 16x16 tile.
#pragma unroll
    for (int j = 0; j < 4; ++j) {
        int gn = tileN + waveN * 64 + j * 16 + l16;
        if (gn >= N) continue;
#pragma unroll
        for (int i = 0; i < 2; ++i) {
#pragma unroll
            for (int r = 0; r < 8; ++r) {
                int gm = tileM + waveM * 32 + i * 16 + half * 8 + r;
                float s = 1.0f;
                if (mode == 1) { int t = target[gm]; s = (t >= 2000  && t < 10000) ? 1.0f : 0.0f; }
                if (mode == 2) { int t = target[gm]; s = (t >= 10000 && t < 50000) ? 1.0f : 0.0f; }
                C[(long long)gm * ldc + gn] = acc[i][j][r] * s;
            }
        }
    }
}

extern "C" void kernel_launch(void* const* d_in, const int* in_sizes, int n_in,
                              void* d_out, int out_size, void* d_ws, size_t ws_size,
                              hipStream_t stream) {
    const float* hidden = (const float*)d_in[0];   // (2048, 1024)
    const int*   target = (const int*)  d_in[1];   // (2048,)
    const float* head_w = (const float*)d_in[2];   // (1024, 2002)
    const float* down0  = (const float*)d_in[3];   // (1024, 1024)
    const float* dec0   = (const float*)d_in[4];   // (1024, 8000)
    const float* down1  = (const float*)d_in[5];   // (1024, 256)
    const float* dec1   = (const float*)d_in[6];   // (256, 40000)

    float* out = (float*)d_out;                    // (2048, 50002) row-major
    float* h0  = (float*)d_ws;                     // (2048, 1024) scratch
    float* h1  = h0 + (size_t)2048 * 1024;         // (2048, 256) scratch

    const int B = 2048, H = 1024;
    const int LDC = 50002;
    dim3 blk(256);

    // h0 = hidden @ down0            (2048x1024x1024)
    adasoft_gemm_bf16_wmma<<<dim3(1024 / BN, B / BM), blk, 0, stream>>>(
        hidden, H, down0, 1024, h0, 1024, B, 1024, H, nullptr, 0);
    // h1 = hidden @ down1            (2048x256x1024)
    adasoft_gemm_bf16_wmma<<<dim3(256 / BN, B / BM), blk, 0, stream>>>(
        hidden, H, down1, 256, h1, 256, B, 256, H, nullptr, 0);
    // head = hidden @ head_w -> out[:, 0:2002]   (ldb=2002 -> scalar-B path)
    adasoft_gemm_bf16_wmma<<<dim3((2002 + BN - 1) / BN, B / BM), blk, 0, stream>>>(
        hidden, H, head_w, 2002, out + 0, LDC, B, 2002, H, nullptr, 0);
    // t0 = (h0 @ dec0) * m0 -> out[:, 2002:10002]   (stream-ordered after h0)
    adasoft_gemm_bf16_wmma<<<dim3((8000 + BN - 1) / BN, B / BM), blk, 0, stream>>>(
        h0, 1024, dec0, 8000, out + 2002, LDC, B, 8000, 1024, target, 1);
    // t1 = (h1 @ dec1) * m1 -> out[:, 10002:50002]  (stream-ordered after h1)
    adasoft_gemm_bf16_wmma<<<dim3((40000 + BN - 1) / BN, B / BM), blk, 0, stream>>>(
        h1, 256, dec1, 40000, out + 10002, LDC, B, 40000, 256, target, 2);
}
